// CustomMultiheadAttention_15814069584008
// MI455X (gfx1250) — compile-verified, matrix operand order checked by reading, not run on hardware
//
#include <hip/hip_runtime.h>
#include <hip/hip_bf16.h>

typedef __attribute__((ext_vector_type(16))) __bf16 v16bf;
typedef __attribute__((ext_vector_type(8)))  __bf16 bhalf8;
typedef __attribute__((ext_vector_type(8)))  float  v8f;
typedef __attribute__((ext_vector_type(4)))  int    i32x4;

#define D_MODEL   1024
#define NUM_HEADS 16
#define HEAD_DIM  64
#define SEQ       2048
#define BATCH     4
#define M_TOTAL   (BATCH * SEQ)   // 8192
#define NEGINF    (-3.0e38f)

// ---- async global->LDS staging (gfx1250 ASYNCcnt path), with safe fallback --
#if defined(__has_builtin)
#  if __has_builtin(__builtin_amdgcn_global_load_async_to_lds_b128)
#    define USE_ASYNC_LDS 1
#  endif
#endif
#ifndef USE_ASYNC_LDS
#  define USE_ASYNC_LDS 0
#endif

__device__ __forceinline__ __bf16 f2bf(float f) {
  union { float f; unsigned u; } v; v.f = f;
  unsigned r = v.u + 0x7fffu + ((v.u >> 16) & 1u);  // round-to-nearest-even
  unsigned short h = (unsigned short)(r >> 16);
  __bf16 b; __builtin_memcpy(&b, &h, sizeof(b)); return b;
}

// Copy 16 bytes global -> LDS. Async path writes LDS directly (no VGPR bounce,
// tracked by ASYNCcnt); fallback is load+ds_store.
__device__ __forceinline__ void copy16_to_lds(__bf16* lds_dst, const __bf16* gsrc) {
#if USE_ASYNC_LDS
  __builtin_amdgcn_global_load_async_to_lds_b128(
      (__attribute__((address_space(1))) i32x4*)(unsigned long long)gsrc,
      (__attribute__((address_space(3))) i32x4*)(unsigned long long)lds_dst,
      0, 0);
#else
  *reinterpret_cast<bhalf8*>(lds_dst) = *reinterpret_cast<const bhalf8*>(gsrc);
#endif
}

__device__ __forceinline__ void lds_copy_fence() {
#if USE_ASYNC_LDS
#  if __has_builtin(__builtin_amdgcn_s_wait_asynccnt)
  __builtin_amdgcn_s_wait_asynccnt(0);
#  else
  asm volatile("s_wait_asynccnt 0" ::: "memory");
#  endif
#endif
}

// Load a 16x32 bf16 WMMA fragment (A-layout; identical pattern serves B when
// the source array is [N,K] row-major): lane holds row (lane&15), k in
// [kb, kb+7] and [kb+16, kb+23] with kb = (lane>>4)*8  (ISA 7.12.2).
__device__ __forceinline__ v16bf load_frag(const __bf16* base, int ld) {
  const int lane = threadIdx.x & 31;
  const int r    = lane & 15;
  const int kb   = (lane >> 4) << 3;
  const __bf16* rowp = base + r * ld + kb;
  bhalf8 lo = *reinterpret_cast<const bhalf8*>(rowp);
  bhalf8 hi = *reinterpret_cast<const bhalf8*>(rowp + 16);
  v16bf f;
#pragma unroll
  for (int i = 0; i < 8; ++i) { f[i] = lo[i]; f[i + 8] = hi[i]; }
  return f;
}

__device__ __forceinline__ v8f wmma_bf16(v16bf a, v16bf b, v8f c) {
  return __builtin_amdgcn_wmma_f32_16x16x32_bf16(
      /*neg_a=*/false, a, /*neg_b=*/false, b,
      /*c_mod=*/(short)0, c, /*reuse_a=*/false, /*reuse_b=*/false);
}

// ---------------------------------------------------------------------------
// fp32 -> bf16 convert
// ---------------------------------------------------------------------------
__global__ void f32_to_bf16_kernel(const float* __restrict__ src,
                                   __bf16* __restrict__ dst, int n) {
  int i = blockIdx.x * blockDim.x + threadIdx.x;
  const int stride = gridDim.x * blockDim.x;
  for (; i < n; i += stride) dst[i] = f2bf(src[i]);
}

// ---------------------------------------------------------------------------
// Tiled bf16 GEMM:  C[M,Nout] = A[M,K] * W[Nout,K]^T   (i.e. y = x @ W.T)
// Block tile 64(M) x 128(N), 8 waves (4x2), wave tile 16x64, K-step 32.
// mode 0: store bf16 as [B,H,N,Hd]  (Q / K projection)
// mode 1: store bf16 as [B,H,Hd,N]  (V projection, transposed)
// mode 2: store fp32 as [M,Nout]    (output projection)
// ---------------------------------------------------------------------------
__global__ __launch_bounds__(256)
void gemm_bf16_kernel(const __bf16* __restrict__ A,
                      const __bf16* __restrict__ W,
                      __bf16* __restrict__ outb,
                      float*  __restrict__ outf,
                      int M, int K, int Nout, int mode) {
  __shared__ __align__(16) __bf16 aT[64 * 32];
  __shared__ __align__(16) __bf16 bT[128 * 32];

  const int tid   = threadIdx.x;
  const int wave  = tid >> 5;
  const int lane  = tid & 31;
  const int waveM = wave & 3;    // 0..3
  const int waveN = wave >> 2;   // 0..1
  const int m0 = blockIdx.x * 64;
  const int n0 = blockIdx.y * 128;

  const v8f vzero = {0.f, 0.f, 0.f, 0.f, 0.f, 0.f, 0.f, 0.f};
  v8f c[4] = {vzero, vzero, vzero, vzero};

  const int ar = tid >> 2, ak = (tid & 3) << 3;   // A coop: 8 bf16/thread
  const int br = tid >> 1, bk = (tid & 1) << 4;   // B coop: 16 bf16/thread

  for (int k0 = 0; k0 < K; k0 += 32) {
    copy16_to_lds(&aT[ar * 32 + ak],     &A[(size_t)(m0 + ar) * K + k0 + ak]);
    copy16_to_lds(&bT[br * 32 + bk],     &W[(size_t)(n0 + br) * K + k0 + bk]);
    copy16_to_lds(&bT[br * 32 + bk + 8], &W[(size_t)(n0 + br) * K + k0 + bk + 8]);
    if (k0 + 32 < K) {  // pull next A tile toward L2 (global_prefetch_b8)
      __builtin_prefetch(&A[(size_t)(m0 + ar) * K + k0 + 32 + ak], 0, 1);
    }
    lds_copy_fence();
    __syncthreads();

    v16bf af = load_frag(&aT[(waveM * 16) * 32], 32);
#pragma unroll
    for (int j = 0; j < 4; ++j) {
      v16bf bf = load_frag(&bT[(waveN * 64 + j * 16) * 32], 32);
      c[j] = wmma_bf16(af, bf, c[j]);
    }
    __syncthreads();
  }

  const int hig = lane >> 4;
  const int col = lane & 15;
#pragma unroll
  for (int j = 0; j < 4; ++j) {
#pragma unroll
    for (int r = 0; r < 8; ++r) {
      const int m = m0 + waveM * 16 + r + hig * 8;
      const int o = n0 + waveN * 64 + j * 16 + col;
      const float val = c[j][r];
      if (mode == 2) {
        outf[(size_t)m * Nout + o] = val;
      } else {
        const int b = m >> 11, tok = m & (SEQ - 1);
        const int h = o >> 6,  d   = o & (HEAD_DIM - 1);
        if (mode == 0)
          outb[(((size_t)(b * NUM_HEADS + h) * SEQ) + tok) * HEAD_DIM + d] = f2bf(val);
        else
          outb[(((size_t)(b * NUM_HEADS + h) * HEAD_DIM) + d) * SEQ + tok] = f2bf(val);
      }
    }
  }
}

// ---------------------------------------------------------------------------
// Flash attention: one block = 128 query rows of one (b,h); 8 waves x 16 rows.
// Streams K/V in 32-key chunks through LDS (shared across waves), online
// softmax with alibi + mask, P re-laid out via LDS for the P*V WMMA.
// Output attn_out bf16 as [B, N, H*Hd] (token-major for the final GEMM).
// ---------------------------------------------------------------------------
__global__ __launch_bounds__(256)
void flash_attn_kernel(const __bf16* __restrict__ Q,    // [B,H,N,Hd]
                       const __bf16* __restrict__ Kt,   // [B,H,N,Hd]
                       const __bf16* __restrict__ Vt,   // [B,H,Hd,N]
                       const float*  __restrict__ alibi,// [1,H,N,N]
                       const int*    __restrict__ mask, // [B,1,N,N]
                       __bf16* __restrict__ attn_out) { // [B,N,D]
  __shared__ __align__(16) __bf16 kT[32 * 64];          // [key][dim]
  __shared__ __align__(16) __bf16 vT[64 * 32];          // [dim][key]
  __shared__ __align__(16) __bf16 pT[8 * 16 * 32];      // per-wave P tile

  const int tid  = threadIdx.x;
  const int wave = tid >> 5;
  const int lane = tid & 31;
  const int hig  = lane >> 4;
  const int col  = lane & 15;
  const int bh = blockIdx.y;
  const int b  = bh >> 4;
  const int h  = bh & 15;
  const int q0 = blockIdx.x * 128 + wave * 16;
  const float SCALE = 0.125f;  // 1/sqrt(64)

  const __bf16* qbase = Q + ((size_t)bh * SEQ + q0) * HEAD_DIM;
  const v16bf aq0 = load_frag(qbase, HEAD_DIM);        // dims 0..31
  const v16bf aq1 = load_frag(qbase + 32, HEAD_DIM);   // dims 32..63

  const v8f vzero = {0.f, 0.f, 0.f, 0.f, 0.f, 0.f, 0.f, 0.f};
  v8f o0 = vzero, o1 = vzero, o2 = vzero, o3 = vzero;
  float mrow[8], lrow[8];
#pragma unroll
  for (int r = 0; r < 8; ++r) { mrow[r] = NEGINF; lrow[r] = 0.f; }

  const int kr = tid >> 3, kd = (tid & 7) << 3;  // K coop load
  const int vr = tid >> 2, vk = (tid & 3) << 3;  // V coop load

  for (int kc = 0; kc < SEQ; kc += 32) {
    copy16_to_lds(&kT[kr * 64 + kd],
                  &Kt[((size_t)bh * SEQ + kc + kr) * HEAD_DIM + kd]);
    copy16_to_lds(&vT[vr * 32 + vk],
                  &Vt[((size_t)bh * HEAD_DIM + vr) * SEQ + kc + vk]);
    lds_copy_fence();
    __syncthreads();

    // S = Q * K^T for two 16-key subtiles (contraction 64 = 2 WMMA each)
    v8f s[2];
#pragma unroll
    for (int sub = 0; sub < 2; ++sub) {
      v8f acc = vzero;
      v16bf bk0 = load_frag(&kT[sub * 16 * 64], 64);
      v16bf bk1 = load_frag(&kT[sub * 16 * 64 + 32], 64);
      acc = wmma_bf16(aq0, bk0, acc);
      acc = wmma_bf16(aq1, bk1, acc);
      s[sub] = acc;
    }

    // online softmax per row (C-fragment: vgpr r + lane-half hig -> row)
#pragma unroll
    for (int r = 0; r < 8; ++r) {
      const int qrow = q0 + r + hig * 8;
      const size_t arow = ((size_t)h * SEQ + qrow) * SEQ + kc;
      const size_t xrow = ((size_t)b * SEQ + qrow) * SEQ + kc;
      float sv0 = s[0][r] * SCALE + alibi[arow + col];
      float sv1 = s[1][r] * SCALE + alibi[arow + 16 + col];
      if (mask[xrow + col] == 0)      sv0 = NEGINF;
      if (mask[xrow + 16 + col] == 0) sv1 = NEGINF;

      float rm = fmaxf(sv0, sv1);            // row-max across 16 lanes
      rm = fmaxf(rm, __shfl_xor(rm, 1));
      rm = fmaxf(rm, __shfl_xor(rm, 2));
      rm = fmaxf(rm, __shfl_xor(rm, 4));
      rm = fmaxf(rm, __shfl_xor(rm, 8));

      const float mn = fmaxf(mrow[r], rm);
      const float al = __expf(mrow[r] - mn);
      const float p0 = __expf(sv0 - mn);
      const float p1 = __expf(sv1 - mn);
      float rs = p0 + p1;                    // row-sum across 16 lanes
      rs += __shfl_xor(rs, 1);
      rs += __shfl_xor(rs, 2);
      rs += __shfl_xor(rs, 4);
      rs += __shfl_xor(rs, 8);

      lrow[r] = lrow[r] * al + rs;
      mrow[r] = mn;
      o0[r] = o0[r] * al; o1[r] = o1[r] * al;
      o2[r] = o2[r] * al; o3[r] = o3[r] * al;

      __bf16* prow = &pT[(wave * 16 + r + hig * 8) * 32];
      prow[col]      = f2bf(p0);
      prow[16 + col] = f2bf(p1);
    }
    __syncthreads();  // make P visible cross-lane before re-load as A-fragment

    const v16bf ap = load_frag(&pT[wave * 16 * 32], 32);   // P 16x32
    o0 = wmma_bf16(ap, load_frag(&vT[0 * 16 * 32], 32), o0);
    o1 = wmma_bf16(ap, load_frag(&vT[1 * 16 * 32], 32), o1);
    o2 = wmma_bf16(ap, load_frag(&vT[2 * 16 * 32], 32), o2);
    o3 = wmma_bf16(ap, load_frag(&vT[3 * 16 * 32], 32), o3);
    __syncthreads();  // before next chunk overwrites kT/vT
  }

#pragma unroll
  for (int r = 0; r < 8; ++r) {
    const float inv = 1.0f / fmaxf(lrow[r], 1e-20f);
    const int qrow = q0 + r + hig * 8;
    const size_t obase = ((size_t)(b * SEQ + qrow)) * D_MODEL + h * HEAD_DIM;
    attn_out[obase +  0 + col] = f2bf(o0[r] * inv);
    attn_out[obase + 16 + col] = f2bf(o1[r] * inv);
    attn_out[obase + 32 + col] = f2bf(o2[r] * inv);
    attn_out[obase + 48 + col] = f2bf(o3[r] * inv);
  }
}

// ---------------------------------------------------------------------------
extern "C" void kernel_launch(void* const* d_in, const int* in_sizes, int n_in,
                              void* d_out, int out_size, void* d_ws, size_t ws_size,
                              hipStream_t stream) {
  const float* x     = (const float*)d_in[0];
  const int*   mask  = (const int*)d_in[1];
  const float* alibi = (const float*)d_in[2];
  const float* Wq    = (const float*)d_in[3];
  const float* Wk    = (const float*)d_in[4];
  const float* Wv    = (const float*)d_in[5];
  const float* Wo    = (const float*)d_in[6];
  float* out = (float*)d_out;

  const size_t N_X = (size_t)M_TOTAL * D_MODEL;   // 8.39M elems
  const size_t N_W = (size_t)D_MODEL * D_MODEL;   // 1.05M elems

  char* ws = (char*)d_ws;
  __bf16* xb    = (__bf16*)ws;                 ws += N_X * 2;
  __bf16* wqb   = (__bf16*)ws;                 ws += N_W * 2;
  __bf16* wkb   = (__bf16*)ws;                 ws += N_W * 2;
  __bf16* wvb   = (__bf16*)ws;                 ws += N_W * 2;
  __bf16* wob   = (__bf16*)ws;                 ws += N_W * 2;
  __bf16* Qb    = (__bf16*)ws;                 ws += N_X * 2;
  __bf16* Kb    = (__bf16*)ws;                 ws += N_X * 2;
  __bf16* Vb    = (__bf16*)ws;                 ws += N_X * 2;
  __bf16* attnb = (__bf16*)ws;                 ws += N_X * 2;

  // converts
  f32_to_bf16_kernel<<<4096, 256, 0, stream>>>(x,  xb,  (int)N_X);
  f32_to_bf16_kernel<<<1024, 256, 0, stream>>>(Wq, wqb, (int)N_W);
  f32_to_bf16_kernel<<<1024, 256, 0, stream>>>(Wk, wkb, (int)N_W);
  f32_to_bf16_kernel<<<1024, 256, 0, stream>>>(Wv, wvb, (int)N_W);
  f32_to_bf16_kernel<<<1024, 256, 0, stream>>>(Wo, wob, (int)N_W);

  // Q/K/V projections
  dim3 ggrid(M_TOTAL / 64, D_MODEL / 128);
  gemm_bf16_kernel<<<ggrid, 256, 0, stream>>>(xb, wqb, Qb, nullptr,
                                              M_TOTAL, D_MODEL, D_MODEL, 0);
  gemm_bf16_kernel<<<ggrid, 256, 0, stream>>>(xb, wkb, Kb, nullptr,
                                              M_TOTAL, D_MODEL, D_MODEL, 0);
  gemm_bf16_kernel<<<ggrid, 256, 0, stream>>>(xb, wvb, Vb, nullptr,
                                              M_TOTAL, D_MODEL, D_MODEL, 1);

  // flash attention
  dim3 agrid(SEQ / 128, BATCH * NUM_HEADS);
  flash_attn_kernel<<<agrid, 256, 0, stream>>>(Qb, Kb, Vb, alibi, mask, attnb);

  // output projection (fp32 result)
  gemm_bf16_kernel<<<ggrid, 256, 0, stream>>>(attnb, wob, nullptr, out,
                                              M_TOTAL, D_MODEL, D_MODEL, 2);
}
